// GatingNetwork_20186346291869
// MI455X (gfx1250) — compile-verified, matrix operand order checked by reading, not run on hardware
//
#include <hip/hip_runtime.h>
#include <hip/hip_bf16.h>
#include <math.h>

typedef _Float16 f16;
typedef __attribute__((ext_vector_type(16))) _Float16 v16h;
typedef __attribute__((ext_vector_type(8)))  _Float16 h8;
typedef __attribute__((ext_vector_type(8)))  float    v8f;
typedef __attribute__((ext_vector_type(4)))  float    f4v;

static constexpr int   kB  = 131072;
static constexpr int   kD  = 256;   // D_IN
static constexpr int   kH  = 256;   // HID
static constexpr int   kNE = 4;
static constexpr int   kCD = 16;
static constexpr float kEps = 1e-5f;

// ---- workspace layout (bytes), all 16B+ aligned ----
static constexpr size_t WS_FCW16 = 0;        // kH*kD f16   = 131072 B
static constexpr size_t WS_WC16  = 131072;   // 64*kD f16   = 32768 B
static constexpr size_t WS_SUM   = 163840;   // 256 f32
static constexpr size_t WS_SUMSQ = 164864;   // 256 f32
static constexpr size_t WS_SCALE = 165888;   // 256 f32
static constexpr size_t WS_SHIFT = 166912;   // 256 f32
static constexpr size_t WS_H16   = 262144;   // kB*kH f16   = 64 MiB (L2-resident)

__device__ __forceinline__ v8f wmma_f16(v16h a, v16h b, v8f c) {
  return __builtin_amdgcn_wmma_f32_16x16x32_f16(false, a, false, b,
                                                (short)0, c, false, false);
}
__device__ __forceinline__ v16h cat16(h8 lo, h8 hi) {
  return __builtin_shufflevector(lo, hi, 0,1,2,3,4,5,6,7,8,9,10,11,12,13,14,15);
}

// ---------------- kernel 0: weight conversion + accumulator zeroing ----------
__global__ __launch_bounds__(256) void prep_kernel(
    const float* __restrict__ fc_w, const float* __restrict__ W_caps,
    f16* __restrict__ fcw16, f16* __restrict__ wc16,
    float* __restrict__ gsum, float* __restrict__ gsumsq) {
  int tid = blockIdx.x * 256 + threadIdx.x;   // 65536 threads
  fcw16[tid] = (f16)fc_w[tid];                // [j][d] row-major, same as fc_w
  if (tid < kNE * kH * kCD) {                 // W_caps flat: e*4096 + d*16 + o
    int e = tid >> 12, rem = tid & 4095;
    int d = rem >> 4, o = rem & 15;
    wc16[(e * kCD + o) * kD + d] = (f16)W_caps[tid];   // [n][d]
  }
  if (tid < kH) { gsum[tid] = 0.f; gsumsq[tid] = 0.f; }
}

// ---------------- kernel 1: h = x @ fc_w^T + b  (+ BN stats, f16 h) ----------
struct Stage { f16 xs[64][40]; f16 wsz[256][40]; };     // padded strides
union ShMem1 { Stage s; f16 hrep[64][264]; };           // phases don't overlap

__global__ __launch_bounds__(256) void gemm1_kernel(
    const float* __restrict__ x, const float* __restrict__ fc_b,
    const f16* __restrict__ fcw16, f16* __restrict__ h16,
    float* __restrict__ gsum, float* __restrict__ gsumsq) {
  __shared__ alignas(16) ShMem1 u;
  __shared__ float csum[256], csumsq[256], fcbs[256];

  const int tid  = threadIdx.x;
  const int wave = tid >> 5, lane = tid & 31, lm = lane & 15;
  const int g    = wave >> 1;              // row-group 0..3 (16 rows each)
  const int wcol = (wave & 1) * 128;       // column half
  const int rb   = blockIdx.x * 64;

  csum[tid] = 0.f; csumsq[tid] = 0.f; fcbs[tid] = fc_b[tid];
  __syncthreads();

  v8f acc[8];
  #pragma unroll
  for (int t = 0; t < 8; ++t)
    #pragma unroll
    for (int j = 0; j < 8; ++j) acc[t][j] = 0.f;

  const int ka = (lane < 16) ? 0 : 8;

  for (int k0 = 0; k0 < kD; k0 += 32) {
    // --- stage x tile (64x32 f32 -> f16, streaming/non-temporal reads) and
    //     fc_w tile (256x32 f16, L2-hot) ---
    {
      int row = tid >> 2, seg = tid & 3;
      const float* src = x + (size_t)(rb + row) * kD + k0 + seg * 8;
      f4v a = __builtin_nontemporal_load((const f4v*)src);        // TH=NT: read-once stream
      f4v b = __builtin_nontemporal_load((const f4v*)(src + 4));  // keep h16 in L2
      h8 hv;
      hv[0]=(f16)a[0]; hv[1]=(f16)a[1]; hv[2]=(f16)a[2]; hv[3]=(f16)a[3];
      hv[4]=(f16)b[0]; hv[5]=(f16)b[1]; hv[6]=(f16)b[2]; hv[7]=(f16)b[3];
      *(h8*)&u.s.xs[row][seg * 8] = hv;

      const f16* wsrc = fcw16 + (size_t)tid * kD + k0;
      #pragma unroll
      for (int q = 0; q < 4; ++q)
        *(h8*)&u.s.wsz[tid][q * 8] = *(const h8*)(wsrc + q * 8);
    }
    __syncthreads();

    // --- WMMA: A = x rows (g*16..+15), B = fc_w columns ---
    const int mrow = g * 16 + lm;
    v16h av = cat16(*(const h8*)&u.s.xs[mrow][ka],
                    *(const h8*)&u.s.xs[mrow][ka + 16]);
    #pragma unroll
    for (int t = 0; t < 8; ++t) {
      int col = wcol + t * 16 + lm;
      v16h bv = cat16(*(const h8*)&u.s.wsz[col][ka],
                      *(const h8*)&u.s.wsz[col][ka + 16]);
      acc[t] = wmma_f16(av, bv, acc[t]);
    }
    __syncthreads();
  }

  // --- epilogue: bias, column stats, repack to f16 ---
  const int rbase = g * 16 + ((lane >> 4) << 3);   // +8 rows for upper half-wave
  #pragma unroll
  for (int t = 0; t < 8; ++t) {
    int col = wcol + t * 16 + lm;
    float bias = fcbs[col];
    float s = 0.f, s2 = 0.f;
    #pragma unroll
    for (int r = 0; r < 8; ++r) {
      float v = acc[t][r] + bias;
      s += v; s2 += v * v;
      u.hrep[rbase + r][col] = (f16)v;
    }
    atomicAdd(&csum[col], s);
    atomicAdd(&csumsq[col], s2);
  }
  __syncthreads();

  // --- coalesced h16 store (RT hint: keep resident in 192MB L2) + stat atomics ---
  {
    int row = tid >> 2, seg = tid & 3;
    f16* dst = h16 + (size_t)(rb + row) * kH + seg * 64;
    const f16* srcl = &u.hrep[row][seg * 64];
    #pragma unroll
    for (int q = 0; q < 8; ++q)
      *(h8*)(dst + q * 8) = *(const h8*)(srcl + q * 8);
    atomicAdd(&gsum[tid], csum[tid]);
    atomicAdd(&gsumsq[tid], csumsq[tid]);
  }
}

// ---------------- kernel 2: finalize BN scale/shift --------------------------
__global__ __launch_bounds__(256) void bnfin_kernel(
    const float* __restrict__ gsum, const float* __restrict__ gsumsq,
    const float* __restrict__ gamma, const float* __restrict__ beta,
    float* __restrict__ scale, float* __restrict__ shift) {
  int j = threadIdx.x;
  float mean = gsum[j] * (1.f / kB);
  float var  = gsumsq[j] * (1.f / kB) - mean * mean;
  float sc   = gamma[j] * rsqrtf(var + kEps);
  scale[j] = sc;
  shift[j] = beta[j] - mean * sc;
}

// ---------------- kernel 3: BN+ReLU -> GEMM2 -> squash/routing/softmax -------
__global__ __launch_bounds__(128) void fused_kernel(
    const f16* __restrict__ h16, const f16* __restrict__ wc16,
    const float* __restrict__ scale, const float* __restrict__ shift,
    const float* __restrict__ out_w, const float* __restrict__ out_b,
    float* __restrict__ out) {
  __shared__ float scl[256], shf[256], owl[64], obl[4];
  __shared__ float ut[4][32][69];     // pad 69: 5*l mod 64 conflict-free

  const int tid = threadIdx.x, wave = tid >> 5, lane = tid & 31;
  const int lm = lane & 15, ka = (lane < 16) ? 0 : 8;
  for (int i = tid; i < 256; i += 128) { scl[i] = scale[i]; shf[i] = shift[i]; }
  if (tid < 64) owl[tid] = out_w[tid];
  if (tid < 4)  obl[tid] = out_b[tid];
  __syncthreads();

  const int r0 = blockIdx.x * 128 + wave * 32;   // 32 rows per wave
  v8f acc[2][4];
  #pragma unroll
  for (int mt = 0; mt < 2; ++mt)
    #pragma unroll
    for (int t = 0; t < 4; ++t)
      #pragma unroll
      for (int j = 0; j < 8; ++j) acc[mt][t][j] = 0.f;

  for (int k0 = 0; k0 < kH; k0 += 32) {
    v16h av[2];
    #pragma unroll
    for (int mt = 0; mt < 2; ++mt) {
      const f16* hp = h16 + (size_t)(r0 + mt * 16 + lm) * kH + k0 + ka;
      h8 lo = *(const h8*)hp;
      h8 hi = *(const h8*)(hp + 16);
      h8 nlo, nhi;
      #pragma unroll
      for (int i = 0; i < 8; ++i) {
        int c0 = k0 + ka + i, c1 = c0 + 16;
        float v0 = fmaf((float)lo[i], scl[c0], shf[c0]); v0 = v0 > 0.f ? v0 : 0.f;
        float v1 = fmaf((float)hi[i], scl[c1], shf[c1]); v1 = v1 > 0.f ? v1 : 0.f;
        nlo[i] = (f16)v0; nhi[i] = (f16)v1;
      }
      av[mt] = cat16(nlo, nhi);
    }
    #pragma unroll
    for (int t = 0; t < 4; ++t) {
      const f16* wp = wc16 + (size_t)(t * 16 + lm) * kD + k0 + ka;
      v16h bv = cat16(*(const h8*)wp, *(const h8*)(wp + 16));
      acc[0][t] = wmma_f16(av[0], bv, acc[0][t]);
      acc[1][t] = wmma_f16(av[1], bv, acc[1][t]);
    }
  }

  // dump u (32 rows x 64) to LDS
  const int rb2 = (lane >> 4) << 3;
  #pragma unroll
  for (int mt = 0; mt < 2; ++mt)
    #pragma unroll
    for (int t = 0; t < 4; ++t)
      #pragma unroll
      for (int r = 0; r < 8; ++r)
        ut[wave][mt * 16 + rb2 + r][t * 16 + lm] = acc[mt][t][r];
  __syncthreads();

  // ---- per-row squash + 3-iter routing (scalar recurrences, N_IN_CAPS==1) ----
  float un2[4], fsc[4], uhn2[4];
  #pragma unroll
  for (int e = 0; e < 4; ++e) {
    float s = 0.f;
    #pragma unroll
    for (int o = 0; o < 16; ++o) {
      float v = ut[wave][lane][e * 16 + o];
      s += v * v;
    }
    un2[e] = s;
    float nrm = sqrtf(s);
    fsc[e]  = nrm / ((1.f + s) * (nrm + 1e-8f));   // squash scale for u
    uhn2[e] = fsc[e] * fsc[e] * s;                 // |u_hat|^2
  }
  float bb[4] = {0.f, 0.f, 0.f, 0.f}, c[4], g[4];
  #pragma unroll
  for (int it = 0; it < 3; ++it) {
    float m = fmaxf(fmaxf(bb[0], bb[1]), fmaxf(bb[2], bb[3]));
    float cs = 0.f;
    #pragma unroll
    for (int e = 0; e < 4; ++e) { c[e] = __expf(bb[e] - m); cs += c[e]; }
    float inv = 1.f / cs;
    #pragma unroll
    for (int e = 0; e < 4; ++e) {
      c[e] *= inv;
      float sn2 = c[e] * c[e] * uhn2[e];
      float sn  = sqrtf(sn2);
      g[e] = sn / ((1.f + sn2) * (sn + 1e-8f));    // squash scale for s
      if (it < 2) bb[e] += c[e] * g[e] * uhn2[e];  // <u_hat, v>
    }
  }

  // ---- logits = (alpha * u) . out_w + b ; softmax over k ; NT store ----
  float* orow = out + (size_t)(r0 + lane) * 16;
  #pragma unroll
  for (int e = 0; e < 4; ++e) {
    float alpha = c[e] * g[e] * fsc[e];            // v = alpha * u
    float lg[4] = {0.f, 0.f, 0.f, 0.f};
    #pragma unroll
    for (int o = 0; o < 16; ++o) {
      float v = ut[wave][lane][e * 16 + o];
      #pragma unroll
      for (int k = 0; k < 4; ++k) lg[k] = fmaf(v, owl[k * 16 + o], lg[k]);
    }
    #pragma unroll
    for (int k = 0; k < 4; ++k) lg[k] = fmaf(alpha, lg[k], obl[k]);
    float m = fmaxf(fmaxf(lg[0], lg[1]), fmaxf(lg[2], lg[3]));
    float ex[4], es = 0.f;
    #pragma unroll
    for (int k = 0; k < 4; ++k) { ex[k] = __expf(lg[k] - m); es += ex[k]; }
    float inv = 1.f / es;
    f4v o4 = {ex[0] * inv, ex[1] * inv, ex[2] * inv, ex[3] * inv};
    __builtin_nontemporal_store(o4, (f4v*)(orow + e * 4));  // written once, never read
  }
}

// ---------------- host launcher ---------------------------------------------
extern "C" void kernel_launch(void* const* d_in, const int* in_sizes, int n_in,
                              void* d_out, int out_size, void* d_ws, size_t ws_size,
                              hipStream_t stream) {
  const float* x      = (const float*)d_in[0];
  const float* fc_w   = (const float*)d_in[1];
  const float* fc_b   = (const float*)d_in[2];
  const float* gamma  = (const float*)d_in[3];
  const float* beta   = (const float*)d_in[4];
  const float* W_caps = (const float*)d_in[5];
  const float* out_w  = (const float*)d_in[6];
  const float* out_b  = (const float*)d_in[7];
  float* out = (float*)d_out;

  char* ws = (char*)d_ws;
  f16*   fcw16  = (f16*)(ws + WS_FCW16);
  f16*   wc16   = (f16*)(ws + WS_WC16);
  float* gsum   = (float*)(ws + WS_SUM);
  float* gsumsq = (float*)(ws + WS_SUMSQ);
  float* scl    = (float*)(ws + WS_SCALE);
  float* shf    = (float*)(ws + WS_SHIFT);
  f16*   h16    = (f16*)(ws + WS_H16);

  prep_kernel<<<256, 256, 0, stream>>>(fc_w, W_caps, fcw16, wc16, gsum, gsumsq);
  gemm1_kernel<<<kB / 64, 256, 0, stream>>>(x, fc_b, fcw16, h16, gsum, gsumsq);
  bnfin_kernel<<<1, 256, 0, stream>>>(gsum, gsumsq, gamma, beta, scl, shf);
  fused_kernel<<<kB / 128, 128, 0, stream>>>(h16, wc16, scl, shf, out_w, out_b, out);
}
